// SelfAttentionHead_19662360281338
// MI455X (gfx1250) — matrix operand order repaired and checked by reading, not compiled, not run on hardware
//
#include <hip/hip_runtime.h>
#include <hip/hip_bf16.h>
#include <math.h>

// Problem constants (B, T, C, H) from the reference
#define BB 4
#define TT 4096
#define CC 2048
#define HH 128

typedef __attribute__((ext_vector_type(16))) __bf16 v16bf;
typedef __attribute__((ext_vector_type(8)))  __bf16 v8bf;
typedef __attribute__((ext_vector_type(8)))  float  v8f;
typedef __attribute__((ext_vector_type(4)))  unsigned int u32x4;
typedef __attribute__((ext_vector_type(8)))  int i32x8;
typedef __attribute__((ext_vector_type(4)))  int i32x4;

#define HAVE_TDM __has_builtin(__builtin_amdgcn_tensor_load_to_lds)

static __device__ __forceinline__ unsigned short f2bf(float f) {
  unsigned int u = __float_as_uint(f);
  u += 0x7FFFu + ((u >> 16) & 1u);           // round to nearest even
  return (unsigned short)(u >> 16);
}

// ---------------------------------------------------------------------------
// WMMA fragment loaders (wave32, v_wmma_f32_16x16x32_bf16 layouts per ISA 7.12.2)
//
// A operand, 16(M) x 32(K) bf16:
//   lane = half*16 + m ; element e<8 : K = half*8 + e ; e>=8 : K = 16 + half*8 + (e-8)
static __device__ __forceinline__ v16bf load_a_frag(const unsigned short* base,
                                                    int ld, int row0, int k0, int lane) {
  const int m = lane & 15, half = lane >> 4;
  const unsigned short* p0 = base + (size_t)(row0 + m) * (size_t)ld + k0 + half * 8;
  v8bf lo = *(const v8bf*)p0;
  v8bf hi = *(const v8bf*)(p0 + 16);
  v16bf r;
#pragma unroll
  for (int e = 0; e < 8; ++e) { r[e] = lo[e]; r[e + 8] = hi[e]; }
  return r;
}

// B operand, 32(K) x 16(N) bf16, source stored N-major (row n, K contiguous):
//   lane = half*16 + n ; element e<8 : K = half*8 + e ; e>=8 : K = 16 + half*8 + (e-8)
static __device__ __forceinline__ v16bf load_b_frag(const unsigned short* __restrict__ base,
                                                    int ld, int n0, int k0, int lane) {
  const int n = lane & 15, half = lane >> 4;
  const unsigned short* p0 = base + (size_t)(n0 + n) * (size_t)ld + k0 + half * 8;
  v8bf lo = *(const v8bf*)p0;
  v8bf hi = *(const v8bf*)(p0 + 16);
  v16bf r;
#pragma unroll
  for (int e = 0; e < 8; ++e) { r[e] = lo[e]; r[e + 8] = hi[e]; }
  return r;
}

static __device__ __forceinline__ v8f wmma_bf16(v16bf a, v16bf b, v8f c) {
  return __builtin_amdgcn_wmma_f32_16x16x32_bf16(false, a, false, b, (short)0, c, false, false);
}

#if HAVE_TDM
// ---------------------------------------------------------------------------
// Tensor Data Mover: async 2-D tile load (global -> LDS), bf16 elements.
// D# built per CDNA5 ISA 8.3/8.4: group0 = {count=1, lds_addr, gaddr, type=2},
// group1 = {data_size=2B, tensor dims, tile dims, dim0 stride}.
static __device__ __forceinline__ void tdm_load_2d(unsigned lds_off,
                                                   const unsigned short* gptr,
                                                   unsigned tile_x, unsigned tile_y,
                                                   unsigned tensor_x, unsigned tensor_y,
                                                   unsigned stride_x /* elements */) {
  unsigned long long ga = (unsigned long long)(size_t)gptr;
  u32x4 g0;
  g0[0] = 1u;                                            // count=1, no gather
  g0[1] = lds_off;                                       // LDS byte address
  g0[2] = (unsigned)(ga & 0xFFFFFFFFu);                  // global_addr[31:0]
  g0[3] = (unsigned)((ga >> 32) & 0x01FFFFFFu) | (2u << 30);  // [56:32] | type=2
  i32x8 g1;
  g1[0] = (int)(1u << 16);                               // data_size=1 (2 bytes)
  g1[1] = (int)((tensor_x & 0xFFFFu) << 16);             // tensor_dim0[15:0]
  g1[2] = (int)((tensor_x >> 16) | ((tensor_y & 0xFFFFu) << 16));
  g1[3] = (int)((tensor_y >> 16) | (tile_x << 16));      // tile_dim0
  g1[4] = (int)(tile_y & 0xFFFFu);                       // tile_dim1 (tile_dim2=0)
  g1[5] = (int)stride_x;                                 // tensor_dim0_stride[31:0]
  g1[6] = 0;
  g1[7] = 0;
  i32x4 z4 = {0, 0, 0, 0};
#if defined(__clang_major__) && (__clang_major__ >= 23)
  i32x8 z8 = {0, 0, 0, 0, 0, 0, 0, 0};
  __builtin_amdgcn_tensor_load_to_lds(g0, g1, z4, z4, z8, 0);
#else
  __builtin_amdgcn_tensor_load_to_lds(g0, g1, z4, z4, 0);
#endif
}
#endif  // HAVE_TDM

// ---------------------------------------------------------------------------
// Kernel 1: fp32 -> bf16 conversion of the activations
__global__ void __launch_bounds__(256) cvt_idx_kernel(const float* __restrict__ x,
                                                      unsigned short* __restrict__ y) {
  const int i = blockIdx.x * 256 + threadIdx.x;      // grid sized exactly: B*T*C threads
  y[i] = f2bf(x[i]);
}

// Kernel 2: fp32 -> bf16 conversion of Wq/Wk/Wv into one [3][H][C] buffer
__global__ void __launch_bounds__(256) cvt_w_kernel(const float* __restrict__ wq,
                                                    const float* __restrict__ wk,
                                                    const float* __restrict__ wv,
                                                    unsigned short* __restrict__ y) {
  const int i = blockIdx.x * 256 + threadIdx.x;      // 3*H*C threads
  const int mat = i >> 18;                           // H*C = 2^18
  const int rem = i & ((1 << 18) - 1);
  const float* src = (mat == 0) ? wq : (mat == 1) ? wk : wv;
  y[i] = f2bf(src[rem]);
}

// ---------------------------------------------------------------------------
// Kernel 3: QKV projection.  out[m][h] = sum_c idx[m][c] * W[h][c]
// One wave computes a 16(M) x 128(H) tile for one of the 3 matrices.
// Activations are streamed into LDS with the Tensor Data Mover (double-buffered
// 16x256 bf16 slabs, TENSORcnt-gated); weights come from global (L2-resident).
// q,k stored [B*T][H] bf16; v stored transposed [B][H][T] bf16.
__global__ void __launch_bounds__(128) qkv_proj_kernel(const unsigned short* __restrict__ idxbf,
                                                       const unsigned short* __restrict__ wbf,
                                                       unsigned short* __restrict__ qbuf,
                                                       unsigned short* __restrict__ kbuf,
                                                       unsigned short* __restrict__ vT) {
  const int lane = threadIdx.x & 31;
  const int wid  = threadIdx.x >> 5;
  const int mt   = blockIdx.x * 4 + wid;             // 0..1023 M-tiles
  const int mat  = blockIdx.y;                       // 0=q 1=k 2=v
  const int row0 = mt * 16;
  const unsigned short* W = wbf + (size_t)mat * HH * CC;

  v8f acc[8] = {};

#if HAVE_TDM
  __shared__ __align__(16) unsigned short As[4][2][16 * 256];   // 64 KB / block
  const unsigned short* Arow = idxbf + (size_t)row0 * CC;
  const unsigned lds0 = (unsigned)(size_t)&As[wid][0][0];
  const unsigned lds1 = (unsigned)(size_t)&As[wid][1][0];

  tdm_load_2d(lds0, Arow,       256, 16, CC, BB * TT, CC);
  tdm_load_2d(lds1, Arow + 256, 256, 16, CC, BB * TT, CC);

  for (int s = 0; s < 8; ++s) {                      // 8 slabs of 256 K each
    if (s < 7) __builtin_amdgcn_s_wait_tensorcnt(1); // oldest slab complete
    else       __builtin_amdgcn_s_wait_tensorcnt(0); // last slab complete
    const unsigned short* Abuf = &As[wid][s & 1][0];
#pragma unroll
    for (int kk = 0; kk < 8; ++kk) {
      v16bf a = load_a_frag(Abuf, 256, 0, kk * 32, lane);       // ds_load gathers
      v16bf bf[8];
#pragma unroll
      for (int nt = 0; nt < 8; ++nt)
        bf[nt] = load_b_frag(W, CC, nt * 16, s * 256 + kk * 32, lane);
#pragma unroll
      for (int nt = 0; nt < 8; ++nt)
        acc[nt] = wmma_bf16(a, bf[nt], acc[nt]);
    }
    if (s + 2 < 8)                                   // refill the slab just drained
      tdm_load_2d((s & 1) ? lds1 : lds0, Arow + (s + 2) * 256,
                  256, 16, CC, BB * TT, CC);
  }
#else
  for (int k0 = 0; k0 < CC; k0 += 32) {
    v16bf a = load_a_frag(idxbf, CC, row0, k0, lane);
    v16bf bf[8];
#pragma unroll
    for (int nt = 0; nt < 8; ++nt) bf[nt] = load_b_frag(W, CC, nt * 16, k0, lane);
#pragma unroll
    for (int nt = 0; nt < 8; ++nt) acc[nt] = wmma_bf16(a, bf[nt], acc[nt]);
  }
#endif

  const int half = lane >> 4, col0 = lane & 15;
#pragma unroll
  for (int nt = 0; nt < 8; ++nt) {
#pragma unroll
    for (int r = 0; r < 8; ++r) {
      const int row = row0 + r + 8 * half;           // global row in [0, B*T)
      const int h   = nt * 16 + col0;
      const unsigned short val = f2bf(acc[nt][r]);
      if (mat == 0) {
        qbuf[(size_t)row * HH + h] = val;
      } else if (mat == 1) {
        kbuf[(size_t)row * HH + h] = val;
      } else {
        const int b = row >> 12;                     // T = 4096
        const int t = row & (TT - 1);
        vT[((size_t)b * HH + h) * TT + t] = val;
      }
    }
  }
}

// ---------------------------------------------------------------------------
// Kernel 4: causal flash attention.
// Reference: wei[i][j] = k_i . q_j  (roles swapped) -> Q' = k, K' = q, V' = v.
// One wave owns a 16-row i-tile; j-blocks of 32 columns.
__global__ void __launch_bounds__(128) attn_kernel(const unsigned short* __restrict__ qbuf,
                                                   const unsigned short* __restrict__ kbuf,
                                                   const unsigned short* __restrict__ vT,
                                                   float* __restrict__ out) {
  __shared__ __align__(16) unsigned short Pl[4][16 * 32];  // per-wave P staging
  const int lane = threadIdx.x & 31;
  const int wid  = threadIdx.x >> 5;
  const int tile = blockIdx.x * 4 + wid;             // 0..1023
  const int b    = tile >> 8;                        // T/16 = 256 tiles per batch
  const int i0   = (tile & 255) * 16;
  const int half = lane >> 4, col0 = lane & 15;

  const unsigned short* Qp = kbuf + (size_t)b * TT * HH;   // attention "queries" = k
  const unsigned short* Kp = qbuf + (size_t)b * TT * HH;   // attention "keys"    = q
  const unsigned short* Vp = vT   + (size_t)b * HH * TT;   // values, [H][T]

  // Q' row fragments for this i-tile: 16 x 128 split into 4 K-chunks of 32
  v16bf aq[4];
#pragma unroll
  for (int kk = 0; kk < 4; ++kk) aq[kk] = load_a_frag(Qp, HH, i0, kk * 32, lane);

  v8f acc[8] = {};
  float mrow[8], lrow[8];
#pragma unroll
  for (int r = 0; r < 8; ++r) { mrow[r] = -__builtin_inff(); lrow[r] = 0.0f; }

  const float scale = 0.02209708691207961f;          // 2048^-0.5

  for (int j0 = 0; j0 <= i0 + 15; j0 += 32) {
    // ---- S = Q' x K'^T : load all 8 B fragments first (one clause), then WMMA
    v16bf bks[8];
#pragma unroll
    for (int kk = 0; kk < 4; ++kk) {
      bks[kk]     = load_b_frag(Kp, HH, j0,      kk * 32, lane);
      bks[kk + 4] = load_b_frag(Kp, HH, j0 + 16, kk * 32, lane);
    }
    v8f s0 = {}, s1 = {};
#pragma unroll
    for (int kk = 0; kk < 4; ++kk) {
      s0 = wmma_bf16(aq[kk], bks[kk],     s0);
      s1 = wmma_bf16(aq[kk], bks[kk + 4], s1);
    }

    // ---- scale + causal mask (only blocks crossing the diagonal need masking)
    const bool need_mask = (j0 + 31 > i0);
#pragma unroll
    for (int r = 0; r < 8; ++r) {
      s0[r] *= scale;
      s1[r] *= scale;
      if (need_mask) {
        const int ig = i0 + r + 8 * half;
        if (j0 + col0      > ig) s0[r] = -__builtin_inff();
        if (j0 + 16 + col0 > ig) s1[r] = -__builtin_inff();
      }
    }

    // ---- online softmax (row = (r, half); columns live across 16 lanes)
#pragma unroll
    for (int r = 0; r < 8; ++r) {
      float mx = fmaxf(s0[r], s1[r]);
      mx = fmaxf(mx, __shfl_xor(mx, 1, 32));
      mx = fmaxf(mx, __shfl_xor(mx, 2, 32));
      mx = fmaxf(mx, __shfl_xor(mx, 4, 32));
      mx = fmaxf(mx, __shfl_xor(mx, 8, 32));
      const float mnew = fmaxf(mrow[r], mx);
      const float corr = __expf(mrow[r] - mnew);     // exp(-inf) = 0 on first block
      mrow[r] = mnew;
      const float p0 = __expf(s0[r] - mnew);         // masked -> exp(-inf) = 0
      const float p1 = __expf(s1[r] - mnew);
      s0[r] = p0; s1[r] = p1;
      float rs = p0 + p1;
      rs += __shfl_xor(rs, 1, 32);
      rs += __shfl_xor(rs, 2, 32);
      rs += __shfl_xor(rs, 4, 32);
      rs += __shfl_xor(rs, 8, 32);
      lrow[r] = lrow[r] * corr + rs;
#pragma unroll
      for (int ht = 0; ht < 8; ++ht) acc[ht][r] *= corr;
    }

    // ---- re-layout P (16x32 f32 C-layout -> bf16 A-layout) via LDS
    unsigned short* P = &Pl[wid][0];
#pragma unroll
    for (int r = 0; r < 8; ++r) {
      const int il = r + 8 * half;
      P[il * 32 + col0]      = f2bf(s0[r]);
      P[il * 32 + col0 + 16] = f2bf(s1[r]);
    }
    // same-wave DS store->load: compiler inserts s_wait_dscnt; no barrier needed
    v8bf plo = *(const v8bf*)&Pl[wid][(lane & 15) * 32 + half * 8];
    v8bf phi = *(const v8bf*)&Pl[wid][(lane & 15) * 32 + 16 + half * 8];
    v16bf ap;
#pragma unroll
    for (int e = 0; e < 8; ++e) { ap[e] = plo[e]; ap[e + 8] = phi[e]; }

    // ---- acc += P x V : load all 8 V fragments, then WMMA chain
    v16bf bvs[8];
#pragma unroll
    for (int ht = 0; ht < 8; ++ht) bvs[ht] = load_b_frag(Vp, TT, ht * 16, j0, lane);
#pragma unroll
    for (int ht = 0; ht < 8; ++ht) acc[ht] = wmma_bf16(ap, bvs[ht], acc[ht]);
  }

  // ---- normalize and store fp32 output [B][T][H]
#pragma unroll
  for (int ht = 0; ht < 8; ++ht) {
#pragma unroll
    for (int r = 0; r < 8; ++r) {
      const int i = i0 + r + 8 * half;
      const int h = ht * 16 + col0;
      out[((size_t)b * TT + i) * HH + h] = acc[ht][r] / lrow[r];
    }
  }
}

// ---------------------------------------------------------------------------
extern "C" void kernel_launch(void* const* d_in, const int* in_sizes, int n_in,
                              void* d_out, int out_size, void* d_ws, size_t ws_size,
                              hipStream_t stream) {
  const float* idx = (const float*)d_in[0];
  const float* Wq  = (const float*)d_in[1];
  const float* Wk  = (const float*)d_in[2];
  const float* Wv  = (const float*)d_in[3];
  float* out = (float*)d_out;

  // Workspace layout (bytes):
  //   idxbf : B*T*C * 2          = 67108864
  //   wbf   : 3*H*C * 2          =  1572864
  //   qbuf  : B*T*H * 2          =  4194304
  //   kbuf  : B*T*H * 2          =  4194304
  //   vT    : B*H*T * 2          =  4194304   (total ~77.5 MB)
  char* ws = (char*)d_ws;
  unsigned short* idxbf = (unsigned short*)(ws);
  unsigned short* wbf   = (unsigned short*)(ws + 67108864);
  unsigned short* qbuf  = (unsigned short*)(ws + 67108864 + 1572864);
  unsigned short* kbuf  = (unsigned short*)(ws + 67108864 + 1572864 + 4194304);
  unsigned short* vT    = (unsigned short*)(ws + 67108864 + 1572864 + 2 * 4194304);

  cvt_idx_kernel<<<(BB * TT * CC) / 256, 256, 0, stream>>>(idx, idxbf);
  cvt_w_kernel<<<(3 * HH * CC) / 256, 256, 0, stream>>>(Wq, Wk, Wv, wbf);

  dim3 gproj((BB * TT / 16) / 4, 3);                 // 256 x 3 blocks, 4 waves each
  qkv_proj_kernel<<<gproj, 128, 0, stream>>>(idxbf, wbf, qbuf, kbuf, vT);

  attn_kernel<<<(BB * TT / 16) / 4, 128, 0, stream>>>(qbuf, kbuf, vT, out);
}
